// EncoderBlock_70085276336760
// MI455X (gfx1250) — matrix operand order repaired and checked by reading, not compiled, not run on hardware
//
#include <hip/hip_runtime.h>
#include <hip/hip_bf16.h>

typedef _Float16 h16;
typedef h16 v16h __attribute__((ext_vector_type(16)));
typedef h16 v8h  __attribute__((ext_vector_type(8)));
typedef float v8f __attribute__((ext_vector_type(8)));
typedef int v4i128 __attribute__((vector_size(16)));   // pointee type for async-LDS builtin

namespace cfg {
constexpr int Bc = 4, Sc = 2048, Dc = 1024, Hc = 16, Fc = 4096, HDc = 64;
constexpr int BS = Bc * Sc;          // 8192 rows
constexpr int QKVN = 3 * Dc;         // 3072
}

#if __has_builtin(__builtin_amdgcn_global_load_async_to_lds_b128)
#define USE_ASYNC_LDS 1
#else
#define USE_ASYNC_LDS 0
#endif

#define LDS_AS __attribute__((address_space(3)))
#define GLB_AS __attribute__((address_space(1)))

// 16-byte global -> LDS copy (async DMA when available)
__device__ __forceinline__ void stage16(const h16* g, h16* l) {
#if USE_ASYNC_LDS
  void* gv = (void*)g;
  void* lv = (void*)l;
  __builtin_amdgcn_global_load_async_to_lds_b128((GLB_AS v4i128*)gv, (LDS_AS v4i128*)lv,
                                                 /*offset=*/0, /*cpol=*/0);
#else
  *(v8h*)l = *(const v8h*)g;
#endif
}

__device__ __forceinline__ void stage_wait() {
#if USE_ASYNC_LDS
#if __has_builtin(__builtin_amdgcn_s_wait_asynccnt)
  __builtin_amdgcn_s_wait_asynccnt(0);
#else
  asm volatile("s_wait_asynccnt 0x0" ::: "memory");
#endif
#endif
}

__device__ __forceinline__ v16h pack16(v8h lo, v8h hi) {
  v16h r;
#pragma unroll
  for (int i = 0; i < 8; ++i) { r[i] = lo[i]; r[i + 8] = hi[i]; }
  return r;
}

// ---------------------------------------------------------------------------
// Conversion / packing kernels (weights stored TRANSPOSED: [N][K] f16)
// ---------------------------------------------------------------------------
__global__ void cvt_f32_to_f16(const float* __restrict__ in, h16* __restrict__ out, int n) {
  int i = blockIdx.x * 256 + threadIdx.x;
  if (i < n) out[i] = (h16)in[i];
}

// in: [R][C] f32 -> out: [C][R] f16 (transpose). Writes coalesced.
__global__ void cvt_transpose_f16(const float* __restrict__ in, h16* __restrict__ out,
                                  int R, int C) {
  int idx = blockIdx.x * 256 + threadIdx.x;
  if (idx >= R * C) return;
  int c = idx / R, r = idx - c * R;
  out[(size_t)c * R + r] = (h16)in[(size_t)r * C + c];
}

// Wq/Wk/Wv (H, D, HD) -> f16 [3072][1024]: row j = sel*1024 + h*64 + e, col d
__global__ void pack_qkv_wt(const float* __restrict__ Wq, const float* __restrict__ Wk,
                            const float* __restrict__ Wv, h16* __restrict__ out) {
  int idx = blockIdx.x * 256 + threadIdx.x;            // 0 .. 3072*1024-1
  int j = idx >> 10, d = idx & 1023;
  const float* W = (j < 1024) ? Wq : ((j < 2048) ? Wk : Wv);
  int jj = j & 1023;
  int h = jj >> 6, e = jj & 63;
  out[idx] = (h16)W[(size_t)h * cfg::Dc * cfg::HDc + (size_t)d * cfg::HDc + e];
}

// ---------------------------------------------------------------------------
// Tiled WMMA GEMM: C[M,N] = A[M,K] (f16 row-major) * Bt[N,K]^T (f16, pre-transposed)
// flags: bit0 = relu, bit1 = store f16 (else f32). bias may be null.
// BM=BN=128, BK=32, 256 thr = 8 waves, wave tile 32(M) x 64(N).
// Double-buffered LDS; async global->LDS staging on CDNA5.
// ---------------------------------------------------------------------------
__global__ __launch_bounds__(256) void gemm_wmma(
    const h16* __restrict__ A, const h16* __restrict__ Bt,
    const float* __restrict__ bias,
    float* __restrict__ Cf, h16* __restrict__ Ch,
    int M, int N, int K, int flags) {
  constexpr int AP = 40;   // padded LDS row stride (halves): 32 + 8
  __shared__ __align__(16) h16 As[2][128 * AP];
  __shared__ __align__(16) h16 Bs[2][128 * AP];

  const int t = threadIdx.x;
  const int lane = t & 31, w = t >> 5;
  const int g = lane >> 4, l15 = lane & 15;
  const int wm = (w & 3) * 32, wn = (w >> 2) * 64;
  const int bm = blockIdx.y * 128, bn = blockIdx.x * 128;

  // cooperative staging coords: row r (0..63, +64), k-chunk kc (8 halves)
  const int r = t >> 2, kc = (t & 3) * 8;
  const h16* gA = A + (size_t)(bm + r) * K + kc;
  const h16* gB = Bt + (size_t)(bn + r) * K + kc;

  auto stage = [&](int kt, int buf) {
    stage16(gA + (size_t)kt * 32, &As[buf][r * AP + kc]);
    stage16(gA + (size_t)kt * 32 + (size_t)64 * K, &As[buf][(r + 64) * AP + kc]);
    stage16(gB + (size_t)kt * 32, &Bs[buf][r * AP + kc]);
    stage16(gB + (size_t)kt * 32 + (size_t)64 * K, &Bs[buf][(r + 64) * AP + kc]);
  };

  v8f acc[2][4];
#pragma unroll
  for (int mf = 0; mf < 2; ++mf)
#pragma unroll
    for (int nf = 0; nf < 4; ++nf) acc[mf][nf] = (v8f)0.0f;

  const int kTiles = K / 32;
  stage(0, 0);
  stage_wait();
  __syncthreads();

  for (int kt = 0; kt < kTiles; ++kt) {
    const int buf = kt & 1;
    if (kt + 1 < kTiles) stage(kt + 1, buf ^ 1);   // prefetch next tile into other buffer

    v16h af[2], bf[4];
#pragma unroll
    for (int mf = 0; mf < 2; ++mf) {
      int row = wm + mf * 16 + l15;
      v8h lo = *(const v8h*)&As[buf][row * AP + 8 * g];
      v8h hi = *(const v8h*)&As[buf][row * AP + 16 + 8 * g];
      af[mf] = pack16(lo, hi);
    }
#pragma unroll
    for (int nf = 0; nf < 4; ++nf) {
      int col = wn + nf * 16 + l15;
      v8h lo = *(const v8h*)&Bs[buf][col * AP + 8 * g];
      v8h hi = *(const v8h*)&Bs[buf][col * AP + 16 + 8 * g];
      bf[nf] = pack16(lo, hi);
    }
#pragma unroll
    for (int mf = 0; mf < 2; ++mf)
#pragma unroll
      for (int nf = 0; nf < 4; ++nf)
        acc[mf][nf] = __builtin_amdgcn_wmma_f32_16x16x32_f16(
            false, af[mf], false, bf[nf], (short)0, acc[mf][nf], false, false);

    stage_wait();
    __syncthreads();
  }

  // ---- epilogue ----
#pragma unroll
  for (int mf = 0; mf < 2; ++mf)
#pragma unroll
    for (int nf = 0; nf < 4; ++nf) {
      int n = bn + wn + nf * 16 + l15;
      float bv = bias ? bias[n] : 0.0f;
#pragma unroll
      for (int i = 0; i < 8; ++i) {
        int m = bm + wm + mf * 16 + i + 8 * g;
        float v = acc[mf][nf][i] + bv;
        if (flags & 1) v = v > 0.0f ? v : 0.0f;
        if (flags & 2) Ch[(size_t)m * N + n] = (h16)v;
        else           Cf[(size_t)m * N + n] = v;
      }
    }
}

// ---------------------------------------------------------------------------
// Flash attention (causal): qkv f16 [BS][3072] (Q|K|V per head 64 cols).
// Block = 8 waves = 128 q rows of one (b,h); key tiles of 32, uniform trip count.
// ---------------------------------------------------------------------------
__global__ __launch_bounds__(256) void attn_wmma(const h16* __restrict__ qkv,
                                                 h16* __restrict__ attn_out) {
  constexpr int KP = 72;   // Kt row stride (halves): 64 + 8 pad
  constexpr int VP = 40;   // Vt row stride: 32 + 8 pad
  constexpr int PP = 40;   // P  row stride: 32 + 8 pad
  __shared__ __align__(16) h16 Kt[32 * KP];
  __shared__ __align__(16) h16 Vt[64 * VP];
  __shared__ __align__(16) h16 Pw[8 * 16 * PP];

  const int t = threadIdx.x, lane = t & 31, w = t >> 5;
  const int g = lane >> 4, l15 = lane & 15;
  const int qtile = blockIdx.x;             // 16 tiles of 128 rows
  const int bh = blockIdx.y;                // B*H = 64
  const int b = bh >> 4, h = bh & 15;
  const int qb = qtile * 128 + w * 16;      // this wave's first q row
  const size_t rs = cfg::QKVN;
  const h16* base = qkv + (size_t)b * cfg::Sc * rs;

  // Q fragments (contraction 64 = 2 chunks of 32)
  v16h qa[2];
  {
    const h16* qp = base + (size_t)(qb + l15) * rs + h * 64;
#pragma unroll
    for (int cc = 0; cc < 2; ++cc) {
      v8h lo = *(const v8h*)(qp + cc * 32 + 8 * g);
      v8h hi = *(const v8h*)(qp + cc * 32 + 16 + 8 * g);
      qa[cc] = pack16(lo, hi);
    }
  }

  v8f acc[4];
#pragma unroll
  for (int nf = 0; nf < 4; ++nf) acc[nf] = (v8f)0.0f;
  float rmax[8], rsum[8];
#pragma unroll
  for (int i = 0; i < 8; ++i) { rmax[i] = -3.0e38f; rsum[i] = 0.0f; }

  const int nkt = qtile * 4 + 4;            // uniform across the block
  for (int kt = 0; kt < nkt; ++kt) {
    __syncthreads();
    {   // stage K (row-major) and V (transposed) tiles of 32 keys
      int key = t >> 3, c8 = (t & 7) * 8;
      const h16* kp = base + (size_t)(kt * 32 + key) * rs + 1024 + h * 64 + c8;
      *(v8h*)&Kt[key * KP + c8] = *(const v8h*)kp;
      const h16* vp = base + (size_t)(kt * 32 + key) * rs + 2048 + h * 64 + c8;
      v8h vv = *(const v8h*)vp;
#pragma unroll
      for (int i = 0; i < 8; ++i) Vt[(c8 + i) * VP + key] = vv[i];
    }
    __syncthreads();

    // scores: S[16x32] = Q(16x64) . K^T(64x32)  -> 2x2 WMMAs
    v8f sc[2];
    sc[0] = (v8f)0.0f; sc[1] = (v8f)0.0f;
#pragma unroll
    for (int cc = 0; cc < 2; ++cc)
#pragma unroll
      for (int j = 0; j < 2; ++j) {
        int n = j * 16 + l15;
        v8h lo = *(const v8h*)&Kt[n * KP + cc * 32 + 8 * g];
        v8h hi = *(const v8h*)&Kt[n * KP + cc * 32 + 16 + 8 * g];
        v16h kb = pack16(lo, hi);
        sc[j] = __builtin_amdgcn_wmma_f32_16x16x32_f16(
            false, qa[cc], false, kb, (short)0, sc[j], false, false);
      }

    // causal mask + online softmax (rows live across 16 lanes)
    float p0[8], p1[8], alpha[8];
#pragma unroll
    for (int i = 0; i < 8; ++i) {
      int srow = qb + i + 8 * g;
      int k0 = kt * 32 + l15, k1 = k0 + 16;
      bool m0 = k0 > srow, m1 = k1 > srow;
      float s0 = m0 ? -3.0e38f : sc[0][i] * 0.125f;
      float s1 = m1 ? -3.0e38f : sc[1][i] * 0.125f;
      float mx = fmaxf(s0, s1);
      mx = fmaxf(mx, __shfl_xor(mx, 1));
      mx = fmaxf(mx, __shfl_xor(mx, 2));
      mx = fmaxf(mx, __shfl_xor(mx, 4));
      mx = fmaxf(mx, __shfl_xor(mx, 8));
      float mnew = fmaxf(rmax[i], mx);
      float a = __expf(rmax[i] - mnew);
      float e0 = m0 ? 0.0f : __expf(s0 - mnew);
      float e1 = m1 ? 0.0f : __expf(s1 - mnew);
      float ts = e0 + e1;
      ts += __shfl_xor(ts, 1);
      ts += __shfl_xor(ts, 2);
      ts += __shfl_xor(ts, 4);
      ts += __shfl_xor(ts, 8);
      rsum[i] = rsum[i] * a + ts;
      rmax[i] = mnew;
      alpha[i] = a;
      p0[i] = e0; p1[i] = e1;
    }

    // rescale acc, spill P (f16) to per-wave LDS for C->A re-layout
    h16* pw = &Pw[w * 16 * PP];
#pragma unroll
    for (int i = 0; i < 8; ++i) {
      int m = i + 8 * g;
      pw[m * PP + l15] = (h16)p0[i];
      pw[m * PP + 16 + l15] = (h16)p1[i];
#pragma unroll
      for (int nf = 0; nf < 4; ++nf) acc[nf][i] *= alpha[i];
    }
    __syncthreads();

    // attn += P(16x32) . V(32x64)  -> 4 WMMAs
    v16h pa;
    {
      v8h lo = *(const v8h*)&pw[l15 * PP + 8 * g];
      v8h hi = *(const v8h*)&pw[l15 * PP + 16 + 8 * g];
      pa = pack16(lo, hi);
    }
#pragma unroll
    for (int nf = 0; nf < 4; ++nf) {
      int n = nf * 16 + l15;
      v8h lo = *(const v8h*)&Vt[n * VP + 8 * g];
      v8h hi = *(const v8h*)&Vt[n * VP + 16 + 8 * g];
      v16h vb = pack16(lo, hi);
      acc[nf] = __builtin_amdgcn_wmma_f32_16x16x32_f16(
          false, pa, false, vb, (short)0, acc[nf], false, false);
    }
  }

  // normalize and write f16 attn output [BS][1024]
#pragma unroll
  for (int i = 0; i < 8; ++i) {
    int srow = qb + i + 8 * g;
    float inv = 1.0f / rsum[i];
    size_t o = (size_t)(b * cfg::Sc + srow) * cfg::Dc + h * 64;
#pragma unroll
    for (int nf = 0; nf < 4; ++nf)
      attn_out[o + nf * 16 + l15] = (h16)(acc[nf][i] * inv);
  }
}

// ---------------------------------------------------------------------------
// Fused residual + LayerNorm over D=1024; one block per row, 256 threads.
// outh may be null.
// ---------------------------------------------------------------------------
__global__ __launch_bounds__(256) void resid_ln(
    const float* __restrict__ a, const float* __restrict__ r,
    const float* __restrict__ gamma, const float* __restrict__ beta,
    float* __restrict__ outf, h16* __restrict__ outh) {
  __shared__ float red0[256];
  __shared__ float red1[256];
  const int row = blockIdx.x, tid = threadIdx.x;
  const float* pa = a + (size_t)row * cfg::Dc;
  const float* pr = r + (size_t)row * cfg::Dc;
  float v[4], s = 0.0f, s2 = 0.0f;
#pragma unroll
  for (int i = 0; i < 4; ++i) {
    v[i] = pa[tid + i * 256] + pr[tid + i * 256];
    s += v[i]; s2 += v[i] * v[i];
  }
  red0[tid] = s; red1[tid] = s2;
  __syncthreads();
  for (int off = 128; off > 0; off >>= 1) {
    if (tid < off) { red0[tid] += red0[tid + off]; red1[tid] += red1[tid + off]; }
    __syncthreads();
  }
  float mu = red0[0] * (1.0f / cfg::Dc);
  float var = red1[0] * (1.0f / cfg::Dc) - mu * mu;
  float rstd = rsqrtf(var + 1e-5f);
#pragma unroll
  for (int i = 0; i < 4; ++i) {
    int c = tid + i * 256;
    float y = (v[i] - mu) * rstd * gamma[c] + beta[c];
    outf[(size_t)row * cfg::Dc + c] = y;
    if (outh) outh[(size_t)row * cfg::Dc + c] = (h16)y;
  }
}

// ---------------------------------------------------------------------------
extern "C" void kernel_launch(void* const* d_in, const int* in_sizes, int n_in,
                              void* d_out, int out_size, void* d_ws, size_t ws_size,
                              hipStream_t stream) {
  (void)in_sizes; (void)n_in; (void)out_size; (void)ws_size;
  using namespace cfg;
  const float* x    = (const float*)d_in[0];
  const float* Wq   = (const float*)d_in[1];
  const float* Wk   = (const float*)d_in[2];
  const float* Wv   = (const float*)d_in[3];
  const float* Wo   = (const float*)d_in[4];
  const float* g1   = (const float*)d_in[5];
  const float* be1  = (const float*)d_in[6];
  const float* w_in = (const float*)d_in[7];
  const float* b_in = (const float*)d_in[8];
  const float* w_out= (const float*)d_in[9];
  const float* b_out= (const float*)d_in[10];
  const float* g2   = (const float*)d_in[11];
  const float* be2  = (const float*)d_in[12];
  float* out = (float*)d_out;

  char* ws = (char*)d_ws;
  size_t o = 0;
  auto alloc = [&](size_t bytes) { size_t r = o; o += (bytes + 255) & ~(size_t)255; return r; };
  const size_t o_xh    = alloc((size_t)BS * Dc * 2);        // x f16
  const size_t o_wqkv  = alloc((size_t)QKVN * Dc * 2);      // QKV weights, transposed [3072][1024]
  const size_t o_woh   = alloc((size_t)Dc * Dc * 2);        // Wo^T   [1024][1024]
  const size_t o_winh  = alloc((size_t)Fc * Dc * 2);        // w_in^T [4096][1024]
  const size_t o_wouth = alloc((size_t)Dc * Fc * 2);        // w_out^T[1024][4096]
  const size_t o_x1f   = alloc((size_t)BS * Dc * 4);        // LN1 out f32
  const size_t o_x1h   = alloc((size_t)BS * Dc * 2);        // LN1 out f16
  const size_t o_proj  = alloc((size_t)BS * Dc * 4);        // attn@Wo f32
  const size_t o_qkv   = alloc((size_t)BS * QKVN * 2);      // QKV f16
  const size_t o_attnh = alloc((size_t)BS * Dc * 2);        // attention out f16
  const size_t o_h1    = o_qkv;                             // reuse: QKV+attn dead by FFN1
  const size_t o_f2    = o_proj;                            // reuse: proj dead by FFN2

  h16* Xh    = (h16*)(ws + o_xh);
  h16* Wqkvt = (h16*)(ws + o_wqkv);
  h16* Wot   = (h16*)(ws + o_woh);
  h16* Wint  = (h16*)(ws + o_winh);
  h16* Woutt = (h16*)(ws + o_wouth);
  float* X1f = (float*)(ws + o_x1f);
  h16* X1h   = (h16*)(ws + o_x1h);
  float* Proj= (float*)(ws + o_proj);
  h16* QKV   = (h16*)(ws + o_qkv);
  h16* Attnh = (h16*)(ws + o_attnh);
  h16* H1    = (h16*)(ws + o_h1);
  float* F2  = (float*)(ws + o_f2);

  // 1) conversions / packing (weights transposed to [N][K])
  cvt_f32_to_f16<<<(BS * Dc) / 256, 256, 0, stream>>>(x, Xh, BS * Dc);
  pack_qkv_wt<<<(QKVN * Dc) / 256, 256, 0, stream>>>(Wq, Wk, Wv, Wqkvt);
  cvt_transpose_f16<<<(Dc * Dc) / 256, 256, 0, stream>>>(Wo, Wot, Dc, Dc);
  cvt_transpose_f16<<<(Dc * Fc) / 256, 256, 0, stream>>>(w_in, Wint, Dc, Fc);
  cvt_transpose_f16<<<(Fc * Dc) / 256, 256, 0, stream>>>(w_out, Woutt, Fc, Dc);

  // 2) QKV projection: [8192,3072] = Xh * Wqkvt^T, store f16
  gemm_wmma<<<dim3(QKVN / 128, BS / 128), 256, 0, stream>>>(
      Xh, Wqkvt, nullptr, nullptr, QKV, BS, QKVN, Dc, /*flags=*/2);

  // 3) causal flash attention -> Attnh [8192,1024] f16
  attn_wmma<<<dim3(Sc / 128, Bc * Hc), 256, 0, stream>>>(QKV, Attnh);

  // 4) output projection: Proj = Attnh * Wot^T, f32
  gemm_wmma<<<dim3(Dc / 128, BS / 128), 256, 0, stream>>>(
      Attnh, Wot, nullptr, Proj, nullptr, BS, Dc, Dc, /*flags=*/0);

  // 5) LN1: x1 = LN(x + Proj), emit f32 + f16
  resid_ln<<<BS, 256, 0, stream>>>(x, Proj, g1, be1, X1f, X1h);

  // 6) FFN1: H1 = relu(X1h * Wint^T + b_in), f16
  gemm_wmma<<<dim3(Fc / 128, BS / 128), 256, 0, stream>>>(
      X1h, Wint, b_in, nullptr, H1, BS, Fc, Dc, /*flags=*/3);

  // 7) FFN2: F2 = H1 * Woutt^T + b_out, f32
  gemm_wmma<<<dim3(Dc / 128, BS / 128), 256, 0, stream>>>(
      H1, Woutt, b_out, F2, nullptr, BS, Dc, Fc, /*flags=*/0);

  // 8) LN2: out = LN(x1 + F2)
  resid_ln<<<BS, 256, 0, stream>>>(X1f, F2, g2, be2, out, nullptr);
}
